// CrossDeepNetwork_54365696033284
// MI455X (gfx1250) — matrix operand order repaired and checked by reading, not compiled, not run on hardware
//
#include <hip/hip_runtime.h>
#include <hip/hip_bf16.h>
#include <math.h>

// ---------------- types for WMMA ----------------
typedef __attribute__((ext_vector_type(16))) __bf16 v16bf;
typedef __attribute__((ext_vector_type(8)))  float  v8f;

// ---------------- model constants ----------------
#define BATCH   8192
#define FDIM    1725      // 8*106 + 10*60 + 8*33 + 13
#define KPAD1   1728      // FDIM padded to mult of 32
#define N1      1024
#define N2      512
#define N3      256
#define EPSF    1e-5f
#define LDSROW  40        // 32 K bf16 + 8 pad (80B pitch, 16B aligned)

// ---------------- helpers ----------------
__device__ __forceinline__ unsigned short f2bf(float f) {
    unsigned int u = __float_as_uint(f);
    unsigned int r = u + 0x7FFFu + ((u >> 16) & 1u);   // round-to-nearest-even
    return (unsigned short)(r >> 16);
}
__device__ __forceinline__ float bf2f(unsigned short h) {
    return __uint_as_float(((unsigned int)h) << 16);
}

// async memory->LDS 16B copy (CDNA5 GLOBAL_LOAD_ASYNC_TO_LDS_B128, ASYNCcnt)
#define ASYNC_B128(ldsoff, gptr)                                      \
    asm volatile("global_load_async_to_lds_b128 %0, %1, off"          \
                 :: "v"(ldsoff), "v"((unsigned long long)(gptr))      \
                 : "memory")
#define WAIT_ASYNC() asm volatile("s_wait_asynccnt 0x0" ::: "memory")

// ================= zero =================
__global__ void zero_f_kernel(float* __restrict__ p, int n) {
    int i = blockIdx.x * blockDim.x + threadIdx.x;
    if (i < n) p[i] = 0.0f;
}

// ============ embedding gather + concat -> h (B x F, fp32) ============
__global__ void embed_concat_kernel(const int* __restrict__ sparse,
                                    const float* __restrict__ dense,
                                    const float* __restrict__ ea,
                                    const float* __restrict__ eb,
                                    const float* __restrict__ ec,
                                    float* __restrict__ h)
{
    const long long total = (long long)BATCH * FDIM;
    for (long long i = (long long)blockIdx.x * blockDim.x + threadIdx.x;
         i < total; i += (long long)gridDim.x * blockDim.x) {
        int b   = (int)(i / FDIM);
        int col = (int)(i - (long long)b * FDIM);
        float v;
        if (col < 848) {
            int f = col / 106, d = col - f * 106;
            int idx = sparse[b * 26 + f];
            v = ea[((size_t)f * 100000 + idx) * 106 + d];
        } else if (col < 1448) {
            int c = col - 848; int f = c / 60, d = c - f * 60;
            int idx = sparse[b * 26 + 8 + f];
            v = eb[((size_t)f * 10000 + idx) * 60 + d];
        } else if (col < 1712) {
            int c = col - 1448; int f = c / 33, d = c - f * 33;
            int idx = sparse[b * 26 + 18 + f];
            v = ec[((size_t)f * 1000 + idx) * 33 + d];
        } else {
            v = dense[b * 13 + (col - 1712)];
        }
        h[i] = v;
    }
}

// ============ column sums for batch-norm (atomics over row slices) ============
__global__ void col_stats_kernel(const float* __restrict__ X, int M, int N,
                                 float* __restrict__ sum, float* __restrict__ sumsq,
                                 int rowsPerBlock)
{
    int col = blockIdx.x * blockDim.x + threadIdx.x;
    if (col >= N) return;
    int r0 = blockIdx.y * rowsPerBlock;
    int r1 = r0 + rowsPerBlock; if (r1 > M) r1 = M;
    float s = 0.0f, sq = 0.0f;
    for (int r = r0; r < r1; ++r) {
        float v = X[(size_t)r * N + col];
        s += v; sq += v * v;
    }
    atomicAdd(&sum[col],   s);
    atomicAdd(&sumsq[col], sq);
}

__global__ void finalize_stats_kernel(const float* __restrict__ sum,
                                      const float* __restrict__ sumsq,
                                      float* __restrict__ mean,
                                      float* __restrict__ invstd,
                                      int N, float Mf)
{
    int i = blockIdx.x * blockDim.x + threadIdx.x;
    if (i >= N) return;
    float m = sum[i] / Mf;
    float v = sumsq[i] / Mf - m * m;
    mean[i]   = m;
    invstd[i] = rsqrtf(v + EPSF);
}

// ============ BN(h) in place + bf16 A0 (B x KPAD1, zero tail) ============
__global__ void bn_h_kernel(float* __restrict__ h,
                            const float* __restrict__ mean,
                            const float* __restrict__ invstd,
                            unsigned short* __restrict__ A0)
{
    const long long total = (long long)BATCH * KPAD1;
    for (long long i = (long long)blockIdx.x * blockDim.x + threadIdx.x;
         i < total; i += (long long)gridDim.x * blockDim.x) {
        int b   = (int)(i / KPAD1);
        int col = (int)(i - (long long)b * KPAD1);
        unsigned short o = 0;
        if (col < FDIM) {
            size_t hi = (size_t)b * FDIM + col;
            float v = (h[hi] - mean[col]) * invstd[col];
            h[hi] = v;
            o = f2bf(v);
        }
        A0[i] = o;
    }
}

// ============ BN + ReLU + bf16 convert (GEMM epilogue stage) ============
__global__ void bn_relu_bf16_kernel(const float* __restrict__ Y,
                                    const float* __restrict__ mean,
                                    const float* __restrict__ invstd,
                                    unsigned short* __restrict__ Abf,
                                    long long total, int N)
{
    for (long long i = (long long)blockIdx.x * blockDim.x + threadIdx.x;
         i < total; i += (long long)gridDim.x * blockDim.x) {
        int col = (int)(i % N);
        float v = (Y[i] - mean[col]) * invstd[col];
        v = fmaxf(v, 0.0f);
        Abf[i] = f2bf(v);
    }
}

// ============ fp32 weights (K x N) -> bf16 TRANSPOSED (N x Kpad, zero pad) ============
__global__ void convert_w_t_kernel(const float* __restrict__ W,
                                   unsigned short* __restrict__ Wt,
                                   int K, int Kpad, int N)
{
    const long long total = (long long)N * Kpad;
    for (long long i = (long long)blockIdx.x * blockDim.x + threadIdx.x;
         i < total; i += (long long)gridDim.x * blockDim.x) {
        int n = (int)(i / Kpad);
        int k = (int)(i - (long long)n * Kpad);
        Wt[i] = (k < K) ? f2bf(W[(size_t)k * N + n]) : (unsigned short)0;
    }
}

// ============ bf16 WMMA GEMM: C(MxN,f32) = A(MxK,bf16) * Bt(NxK,bf16)^T + bias ============
// block = 256 threads = 8 waves; tile 128x128x32; wave computes 32x64 (2x4 subtiles)
// Double-buffered LDS filled by GLOBAL_LOAD_ASYNC_TO_LDS_B128 (ASYNCcnt),
// L2 warmed two tiles ahead with global_prefetch_b8.
__global__ void __launch_bounds__(256)
gemm_bf16_kernel(const unsigned short* __restrict__ A,
                 const unsigned short* __restrict__ Bt,
                 const float* __restrict__ bias,
                 float* __restrict__ C,
                 int M, int N, int K)
{
    __shared__ __align__(16) unsigned short As[2][128 * LDSROW];
    __shared__ __align__(16) unsigned short Bs[2][128 * LDSROW];

    const int tid  = threadIdx.x;
    const int lane = tid & 31;
    const int wave = tid >> 5;
    const int wm   = wave & 3;   // 4 wave rows -> 32 M each
    const int wn   = wave >> 2;  // 2 wave cols -> 64 N each
    const int l16  = lane & 15;
    const int half = lane >> 4;

    const int rowBase = blockIdx.y * 128;
    const int colBase = blockIdx.x * 128;

    // staging coordinates: thread owns 16B chunks (row rA and rA+64, K-chunk kc)
    const int rA = tid >> 2;
    const int kc = (tid & 3) * 8;

    const unsigned short* gA0 = A  + (size_t)(rowBase + rA)      * K + kc;
    const unsigned short* gA1 = A  + (size_t)(rowBase + rA + 64) * K + kc;
    const unsigned short* gB0 = Bt + (size_t)(colBase + rA)      * K + kc;
    const unsigned short* gB1 = Bt + (size_t)(colBase + rA + 64) * K + kc;

    unsigned ldsA[2], ldsB[2];
    #pragma unroll
    for (int bi = 0; bi < 2; ++bi) {
        ldsA[bi] = (unsigned)(unsigned long long)&As[bi][rA * LDSROW + kc];
        ldsB[bi] = (unsigned)(unsigned long long)&Bs[bi][rA * LDSROW + kc];
    }
    const unsigned hiOff = 64 * LDSROW * 2; // byte offset of rows 64..127

    auto stage = [&](int buf, int koff) {
        ASYNC_B128(ldsA[buf],         gA0 + koff);
        ASYNC_B128(ldsA[buf] + hiOff, gA1 + koff);
        ASYNC_B128(ldsB[buf],         gB0 + koff);
        ASYNC_B128(ldsB[buf] + hiOff, gB1 + koff);
    };

    v8f acc[2][4];
    {
        v8f z = {0.f,0.f,0.f,0.f,0.f,0.f,0.f,0.f};
        #pragma unroll
        for (int i = 0; i < 2; ++i)
            #pragma unroll
            for (int j = 0; j < 4; ++j) acc[i][j] = z;
    }

    const int nk = K / 32;

    stage(0, 0);
    WAIT_ASYNC();
    __syncthreads();

    for (int t = 0; t < nk; ++t) {
        const int cur = t & 1;
        if (t + 1 < nk) stage(cur ^ 1, (t + 1) * 32);     // LDS fill, one tile ahead
        if (t + 2 < nk) {                                 // L2 warm, two tiles ahead
            __builtin_prefetch(gA0 + (t + 2) * 32, 0, 1);
            __builtin_prefetch(gB0 + (t + 2) * 32, 0, 1);
        }

        union Frag { v16bf v; uint4 q[2]; };
        Frag a[2], b[4];
        const unsigned short* Ab = &As[cur][0];
        const unsigned short* Bb = &Bs[cur][0];
        // A frag (16-bit 16x32 layout): m=lane%16, K halves at 16B offsets 16h / 32+16h
        #pragma unroll
        for (int mi = 0; mi < 2; ++mi) {
            int r = wm * 32 + mi * 16 + l16;
            a[mi].q[0] = *reinterpret_cast<const uint4*>(Ab + r * LDSROW + half * 8);
            a[mi].q[1] = *reinterpret_cast<const uint4*>(Ab + r * LDSROW + 16 + half * 8);
        }
        // B frag (32x16): n=lane%16, K halves at 16B offsets 16h / 32+16h (same as A)
        #pragma unroll
        for (int ni = 0; ni < 4; ++ni) {
            int n = wn * 64 + ni * 16 + l16;
            b[ni].q[0] = *reinterpret_cast<const uint4*>(Bb + n * LDSROW + half * 8);
            b[ni].q[1] = *reinterpret_cast<const uint4*>(Bb + n * LDSROW + 16 + half * 8);
        }
        #pragma unroll
        for (int mi = 0; mi < 2; ++mi)
            #pragma unroll
            for (int ni = 0; ni < 4; ++ni)
                acc[mi][ni] = __builtin_amdgcn_wmma_f32_16x16x32_bf16(
                    false, a[mi].v, false, b[ni].v,
                    (short)0, acc[mi][ni], false, false);

        if (t + 1 < nk) WAIT_ASYNC();   // publish next buffer before barrier
        __syncthreads();
    }

    // ---- epilogue: C layout (m = r + 8*half, n = lane%16) + bias ----
    #pragma unroll
    for (int mi = 0; mi < 2; ++mi) {
        #pragma unroll
        for (int ni = 0; ni < 4; ++ni) {
            const int col = colBase + wn * 64 + ni * 16 + l16;
            const float bv = bias[col];
            float* p = C + (size_t)(rowBase + wm * 32 + mi * 16 + half * 8) * N + col;
            #pragma unroll
            for (int r = 0; r < 8; ++r) { *p = acc[mi][ni][r] + bv; p += N; }
        }
    }
}

// ============ cross network (4 layers) + concat + Wout dot + sigmoid ============
__global__ void __launch_bounds__(256)
cross_final_kernel(const float* __restrict__ hN,          // normalized h, B x F
                   const float* __restrict__ crossW,      // 4 x F
                   const float* __restrict__ crossB,      // 4 x F
                   const unsigned short* __restrict__ d3, // B x 256 bf16
                   const float* __restrict__ Wout,        // F + 256
                   float* __restrict__ out)
{
    __shared__ float x0[FDIM];
    __shared__ float xi[FDIM];
    __shared__ float red[256];

    const int b   = blockIdx.x;
    const int tid = threadIdx.x;
    const float* row = hN + (size_t)b * FDIM;

    for (int c = tid; c < FDIM; c += 256) { float v = row[c]; x0[c] = v; xi[c] = v; }
    __syncthreads();

    for (int l = 0; l < 4; ++l) {
        float p = 0.0f;
        for (int c = tid; c < FDIM; c += 256) p += xi[c] * crossW[l * FDIM + c];
        red[tid] = p; __syncthreads();
        for (int s = 128; s > 0; s >>= 1) {
            if (tid < s) red[tid] += red[tid + s];
            __syncthreads();
        }
        float dot = red[0];
        for (int c = tid; c < FDIM; c += 256)
            xi[c] = x0[c] * dot + crossB[l * FDIM + c] + xi[c];
        __syncthreads();
    }

    float p = 0.0f;
    for (int c = tid; c < FDIM; c += 256) p += xi[c] * Wout[c];
    p += bf2f(d3[(size_t)b * 256 + tid]) * Wout[FDIM + tid];
    red[tid] = p; __syncthreads();
    for (int s = 128; s > 0; s >>= 1) {
        if (tid < s) red[tid] += red[tid + s];
        __syncthreads();
    }
    if (tid == 0) out[b] = 1.0f / (1.0f + expf(-red[0]));
}

// ================== launch ==================
extern "C" void kernel_launch(void* const* d_in, const int* in_sizes, int n_in,
                              void* d_out, int out_size, void* d_ws, size_t ws_size,
                              hipStream_t stream) {
    (void)in_sizes; (void)n_in; (void)out_size; (void)ws_size;

    const int*   sparse = (const int*)  d_in[0];
    const float* dense  = (const float*)d_in[1];
    const float* ea     = (const float*)d_in[2];
    const float* eb     = (const float*)d_in[3];
    const float* ec     = (const float*)d_in[4];
    const float* crossW = (const float*)d_in[5];
    const float* crossB = (const float*)d_in[6];
    const float* W1     = (const float*)d_in[7];
    const float* b1     = (const float*)d_in[8];
    const float* W2     = (const float*)d_in[9];
    const float* b2     = (const float*)d_in[10];
    const float* W3     = (const float*)d_in[11];
    const float* b3     = (const float*)d_in[12];
    const float* Wout   = (const float*)d_in[13];
    float* out = (float*)d_out;

    unsigned char* ws = (unsigned char*)d_ws;
    size_t off = 0;
    auto take = [&](size_t bytes) { size_t o = off; off += (bytes + 255) & ~(size_t)255; return o; };

    float*          h   = (float*)         (ws + take((size_t)BATCH * FDIM  * 4));
    unsigned short* A0  = (unsigned short*)(ws + take((size_t)BATCH * KPAD1 * 2));
    unsigned short* Wt1 = (unsigned short*)(ws + take((size_t)N1 * KPAD1    * 2));
    unsigned short* Wt2 = (unsigned short*)(ws + take((size_t)N2 * N1       * 2));
    unsigned short* Wt3 = (unsigned short*)(ws + take((size_t)N3 * N2       * 2));
    float*          y1  = (float*)         (ws + take((size_t)BATCH * N1    * 4));
    unsigned short* a1  = (unsigned short*)(ws + take((size_t)BATCH * N1    * 2));
    float*          y2  = (float*)         (ws + take((size_t)BATCH * N2    * 4));
    unsigned short* a2  = (unsigned short*)(ws + take((size_t)BATCH * N2    * 2));
    float*          y3  = (float*)         (ws + take((size_t)BATCH * N3    * 4));
    unsigned short* a3  = (unsigned short*)(ws + take((size_t)BATCH * N3    * 2));
    float*          st  = (float*)         (ws + take((size_t)4 * 4 * 2048  * 4));

    auto SUM = [&](int s){ return st + (size_t)s * 4 * 2048 + 0;    };
    auto SSQ = [&](int s){ return st + (size_t)s * 4 * 2048 + 2048; };
    auto MEA = [&](int s){ return st + (size_t)s * 4 * 2048 + 4096; };
    auto ISD = [&](int s){ return st + (size_t)s * 4 * 2048 + 6144; };

    const float Mf = (float)BATCH;

    // 0) zero stats region
    zero_f_kernel<<<(4 * 4 * 2048 + 255) / 256, 256, 0, stream>>>(st, 4 * 4 * 2048);

    // 1) embedding gather + concat
    embed_concat_kernel<<<4096, 256, 0, stream>>>(sparse, dense, ea, eb, ec, h);

    // 2) BN stats on h, finalize, normalize (+ bf16 A0)
    col_stats_kernel<<<dim3((FDIM + 255) / 256, 32), 256, 0, stream>>>(h, BATCH, FDIM, SUM(0), SSQ(0), 256);
    finalize_stats_kernel<<<(FDIM + 255) / 256, 256, 0, stream>>>(SUM(0), SSQ(0), MEA(0), ISD(0), FDIM, Mf);
    bn_h_kernel<<<4096, 256, 0, stream>>>(h, MEA(0), ISD(0), A0);

    // 3) weights -> bf16, transposed (N x Kpad)
    convert_w_t_kernel<<<2048, 256, 0, stream>>>(W1, Wt1, FDIM, KPAD1, N1);
    convert_w_t_kernel<<<1024, 256, 0, stream>>>(W2, Wt2, N1, N1, N2);
    convert_w_t_kernel<<<512,  256, 0, stream>>>(W3, Wt3, N2, N2, N3);

    // 4) layer 1: GEMM + BN + ReLU
    gemm_bf16_kernel<<<dim3(N1 / 128, BATCH / 128), 256, 0, stream>>>(A0, Wt1, b1, y1, BATCH, N1, KPAD1);
    col_stats_kernel<<<dim3(N1 / 256, 32), 256, 0, stream>>>(y1, BATCH, N1, SUM(1), SSQ(1), 256);
    finalize_stats_kernel<<<(N1 + 255) / 256, 256, 0, stream>>>(SUM(1), SSQ(1), MEA(1), ISD(1), N1, Mf);
    bn_relu_bf16_kernel<<<2048, 256, 0, stream>>>(y1, MEA(1), ISD(1), a1, (long long)BATCH * N1, N1);

    // 5) layer 2
    gemm_bf16_kernel<<<dim3(N2 / 128, BATCH / 128), 256, 0, stream>>>(a1, Wt2, b2, y2, BATCH, N2, N1);
    col_stats_kernel<<<dim3(N2 / 256, 32), 256, 0, stream>>>(y2, BATCH, N2, SUM(2), SSQ(2), 256);
    finalize_stats_kernel<<<(N2 + 255) / 256, 256, 0, stream>>>(SUM(2), SSQ(2), MEA(2), ISD(2), N2, Mf);
    bn_relu_bf16_kernel<<<2048, 256, 0, stream>>>(y2, MEA(2), ISD(2), a2, (long long)BATCH * N2, N2);

    // 6) layer 3
    gemm_bf16_kernel<<<dim3(N3 / 128, BATCH / 128), 256, 0, stream>>>(a2, Wt3, b3, y3, BATCH, N3, N2);
    col_stats_kernel<<<dim3(1, 32), 256, 0, stream>>>(y3, BATCH, N3, SUM(3), SSQ(3), 256);
    finalize_stats_kernel<<<1, 256, 0, stream>>>(SUM(3), SSQ(3), MEA(3), ISD(3), N3, Mf);
    bn_relu_bf16_kernel<<<2048, 256, 0, stream>>>(y3, MEA(3), ISD(3), a3, (long long)BATCH * N3, N3);

    // 7) cross network + final dot + sigmoid
    cross_final_kernel<<<BATCH, 256, 0, stream>>>(h, crossW, crossB, a3, Wout, out);
}
